// gca_rince_19902878449980
// MI455X (gfx1250) — matrix operand order repaired and checked by reading, not compiled, not run on hardware
//
#include <hip/hip_runtime.h>

#define N 8192
#define D 128
#define EPSILON 0.05f
// Rows of K kept L2-resident (regular-temporal): 5632 rows * 32KB = 176 MB < 192 MB L2.
// Tail rows are streamed with non-temporal policy so they never evict the head.
#define CACHE_ROWS 5632

typedef __attribute__((ext_vector_type(16))) _Float16 v16h;
typedef __attribute__((ext_vector_type(8)))  _Float16 v8h;
typedef __attribute__((ext_vector_type(8)))  float    v8f;
typedef __attribute__((ext_vector_type(4)))  float    v4f;

static __device__ __forceinline__ v16h cat8(v8h lo, v8h hi) {
  return __builtin_shufflevector(lo, hi, 0,1,2,3,4,5,6,7,8,9,10,11,12,13,14,15);
}

static __device__ __forceinline__ v8f wmma_f16(v16h a, v16h b, v8f c) {
  // D = A(16x32,f16) x B(32x16,f16) + C(16x16,f32)
  return __builtin_amdgcn_wmma_f32_16x16x32_f16(false, a, false, b, (short)0, c,
                                                false, false);
}

// ---------------------------------------------------------------- init
__global__ __launch_bounds__(256) void init_k(float* u, float* v,
                                              unsigned* minbits, float invN) {
  int i = blockIdx.x * 256 + threadIdx.x;
  if (i < N) { u[i] = invN; v[i] = invN; }
  if (i == 0) { *minbits = 0x7F800000u; }  // +inf
}

// ------------------------------------------------- row-normalize -> f16
__global__ __launch_bounds__(128) void normalize_k(const float* __restrict__ z,
                                                   _Float16* __restrict__ zh) {
  __shared__ float red[128];
  const int row = blockIdx.x;
  const int t   = threadIdx.x;
  float val = z[(size_t)row * D + t];
  red[t] = val * val;
  __syncthreads();
  for (int off = 64; off > 0; off >>= 1) {
    if (t < off) red[t] += red[t + off];
    __syncthreads();
  }
  float rn = 1.0f / fmaxf(sqrtf(red[0]), 1e-12f);
  zh[(size_t)row * D + t] = (_Float16)(val * rn);
}

// ------------------------------------- scores = zn @ zn^T via WMMA f16
// Each wave: one 32x32 output block = 2x2 tiles of 16x16, K=128 in 4 chunks.
__global__ __launch_bounds__(256) void gemm_scores(const _Float16* __restrict__ Ah,
                                                   float* __restrict__ S,
                                                   unsigned* __restrict__ minbits) {
  const int lane = threadIdx.x & 31;
  const int wave = threadIdx.x >> 5;
  const int task = blockIdx.x * 8 + wave;   // 0 .. 65535
  const int tI   = task >> 8;               // N/32 = 256 supertiles
  const int tJ   = task & 255;
  const int i0 = tI * 32, j0 = tJ * 32;
  const int hs = lane >> 4;                 // half-wave select
  const int r  = lane & 15;

  const _Float16* arow0 = Ah + (size_t)(i0 + r)      * D;
  const _Float16* arow1 = Ah + (size_t)(i0 + 16 + r) * D;
  const _Float16* brow0 = Ah + (size_t)(j0 + r)      * D;
  const _Float16* brow1 = Ah + (size_t)(j0 + 16 + r) * D;

  v8f c00 = {}; v8f c01 = {}; v8f c10 = {}; v8f c11 = {};

#pragma unroll
  for (int kc = 0; kc < D; kc += 32) {
    // A fragment: lanes 0-15 V0-3 hold K kc+0..7, V4-7 hold kc+16..23;
    // lanes 16-31 shifted by +8.  Two aligned 16B loads per fragment.
    v16h a0 = cat8(*(const v8h*)(arow0 + kc + hs * 8),
                   *(const v8h*)(arow0 + kc + 16 + hs * 8));
    v16h a1 = cat8(*(const v8h*)(arow1 + kc + hs * 8),
                   *(const v8h*)(arow1 + kc + 16 + hs * 8));
    // B fragment: same K layout as A (lanes 0-15 kc+0..7, kc+16..23; lanes 16-31 +8).
    v16h b0 = cat8(*(const v8h*)(brow0 + kc + hs * 8),
                   *(const v8h*)(brow0 + kc + 16 + hs * 8));
    v16h b1 = cat8(*(const v8h*)(brow1 + kc + hs * 8),
                   *(const v8h*)(brow1 + kc + 16 + hs * 8));
    c00 = wmma_f16(a0, b0, c00);
    c01 = wmma_f16(a0, b1, c01);
    c10 = wmma_f16(a1, b0, c10);
    c11 = wmma_f16(a1, b1, c11);
  }

  // C/D layout: VGPR g, lanes 0-15 -> M=g, N=lane; lanes 16-31 -> M=8+g.
  const int colA = j0 + r;
  const int colB = j0 + 16 + r;
  float mv = __builtin_inff();
#pragma unroll
  for (int g = 0; g < 8; ++g) {
    int ra = i0 + hs * 8 + g;
    int rb = i0 + 16 + hs * 8 + g;
    // Scores are read exactly once (by gibbs_k) -> non-temporal stores.
    __builtin_nontemporal_store(c00[g], &S[(size_t)ra * N + colA]);
    __builtin_nontemporal_store(c01[g], &S[(size_t)ra * N + colB]);
    __builtin_nontemporal_store(c10[g], &S[(size_t)rb * N + colA]);
    __builtin_nontemporal_store(c11[g], &S[(size_t)rb * N + colB]);
    mv = fminf(mv, fmaxf(c00[g], 1e-7f));
    mv = fminf(mv, fmaxf(c01[g], 1e-7f));
    mv = fminf(mv, fmaxf(c10[g], 1e-7f));
    mv = fminf(mv, fmaxf(c11[g], 1e-7f));
  }

  // block-reduce min, one atomicMin per block (positive floats: uint order)
  __shared__ float red[256];
  red[threadIdx.x] = mv;
  __syncthreads();
  for (int off = 128; off > 0; off >>= 1) {
    if (threadIdx.x < off) red[threadIdx.x] = fminf(red[threadIdx.x], red[threadIdx.x + off]);
    __syncthreads();
  }
  if (threadIdx.x == 0) atomicMin(minbits, __float_as_uint(red[0]));
}

// ---------------------- K = exp(-cost/eps), diag -> exp(-cmax/eps), in place
// Head rows stored regular-temporal (warm the L2 working set for the Sinkhorn
// passes); tail rows stored non-temporal so they never evict the head.
__global__ __launch_bounds__(256) void gibbs_k(float* __restrict__ K,
                                               const unsigned* __restrict__ minbits) {
  size_t idx = ((size_t)blockIdx.x * 256 + threadIdx.x) * 4;
  const int i = (int)(idx >> 13);      // / 8192
  const int j = (int)(idx & (N - 1));
  const float cmax = 1.0f - __uint_as_float(*minbits);
  v4f s = __builtin_nontemporal_load((const v4f*)(K + idx)); // read-once
  float c0 = (j + 0 == i) ? cmax : 1.0f - fmaxf(s.x, 1e-7f);
  float c1 = (j + 1 == i) ? cmax : 1.0f - fmaxf(s.y, 1e-7f);
  float c2 = (j + 2 == i) ? cmax : 1.0f - fmaxf(s.z, 1e-7f);
  float c3 = (j + 3 == i) ? cmax : 1.0f - fmaxf(s.w, 1e-7f);
  v4f o;
  o.x = expf(-c0 / EPSILON);
  o.y = expf(-c1 / EPSILON);
  o.z = expf(-c2 / EPSILON);
  o.w = expf(-c3 / EPSILON);
  if (i < CACHE_ROWS) {
    *(v4f*)(K + idx) = o;
  } else {
    __builtin_nontemporal_store(o, (v4f*)(K + idx));
  }
}

// ------------------- Sinkhorn step: out[row] = invN / dot(K[row,:], x)
// Head rows: regular loads (L2-resident across all 21 passes).
// Tail rows: non-temporal loads (stream from HBM, do not thrash L2).
__global__ __launch_bounds__(256) void matvec_div(const float* __restrict__ K,
                                                  const float* __restrict__ x,
                                                  float* __restrict__ out,
                                                  float invN) {
  __shared__ float red[256];
  const int row = blockIdx.x;
  const float* Kr = K + (size_t)row * N;
  float s = 0.0f;
  if (row < CACHE_ROWS) {
    for (int j = threadIdx.x * 4; j < N; j += 1024) {
      v4f k4 = *(const v4f*)(Kr + j);
      v4f x4 = *(const v4f*)(x + j);
      s += k4.x * x4.x + k4.y * x4.y + k4.z * x4.z + k4.w * x4.w;
    }
  } else {
    for (int j = threadIdx.x * 4; j < N; j += 1024) {
      v4f k4 = __builtin_nontemporal_load((const v4f*)(Kr + j));
      v4f x4 = *(const v4f*)(x + j);
      s += k4.x * x4.x + k4.y * x4.y + k4.z * x4.z + k4.w * x4.w;
    }
  }
  red[threadIdx.x] = s;
  __syncthreads();
  for (int off = 128; off > 0; off >>= 1) {
    if (threadIdx.x < off) red[threadIdx.x] += red[threadIdx.x + off];
    __syncthreads();
  }
  if (threadIdx.x == 0) out[row] = invN / red[0];
}

// -------- per-row NLL:  nll_i = log(sum_j K_ij v_j) - log K_it - log v_t
__global__ __launch_bounds__(256) void loss_rows(const float* __restrict__ K,
                                                 const float* __restrict__ v,
                                                 float* __restrict__ nll) {
  __shared__ float red[256];
  const int row = blockIdx.x;
  const float* Kr = K + (size_t)row * N;
  float s = 0.0f;
  if (row < CACHE_ROWS) {
    for (int j = threadIdx.x * 4; j < N; j += 1024) {
      v4f k4 = *(const v4f*)(Kr + j);
      v4f v4 = *(const v4f*)(v + j);
      s += k4.x * v4.x + k4.y * v4.y + k4.z * v4.z + k4.w * v4.w;
    }
  } else {
    for (int j = threadIdx.x * 4; j < N; j += 1024) {
      v4f k4 = __builtin_nontemporal_load((const v4f*)(Kr + j));
      v4f v4 = *(const v4f*)(v + j);
      s += k4.x * v4.x + k4.y * v4.y + k4.z * v4.z + k4.w * v4.w;
    }
  }
  red[threadIdx.x] = s;
  __syncthreads();
  for (int off = 128; off > 0; off >>= 1) {
    if (threadIdx.x < off) red[threadIdx.x] += red[threadIdx.x + off];
    __syncthreads();
  }
  if (threadIdx.x == 0) {
    int t = row ^ 1;  // targets: swap adjacent pairs
    nll[row] = logf(red[0]) - logf(Kr[t]) - logf(v[t]);
  }
}

// ---------------------- loss = (mean nll)^Q / Q = 2*sqrt(mean nll)
__global__ __launch_bounds__(256) void finalize_k(const float* __restrict__ nll,
                                                  float* __restrict__ out) {
  __shared__ float red[256];
  float s = 0.0f;
  for (int i = threadIdx.x; i < N; i += 256) s += nll[i];  // fixed order: deterministic
  red[threadIdx.x] = s;
  __syncthreads();
  for (int off = 128; off > 0; off >>= 1) {
    if (threadIdx.x < off) red[threadIdx.x] += red[threadIdx.x + off];
    __syncthreads();
  }
  if (threadIdx.x == 0) out[0] = 2.0f * sqrtf(red[0] / (float)N);
}

extern "C" void kernel_launch(void* const* d_in, const int* in_sizes, int n_in,
                              void* d_out, int out_size, void* d_ws, size_t ws_size,
                              hipStream_t stream) {
  (void)in_sizes; (void)n_in; (void)out_size; (void)ws_size;
  const float* z = (const float*)d_in[0];
  float* out = (float*)d_out;

  char* ws = (char*)d_ws;
  const size_t K_BYTES  = (size_t)N * N * sizeof(float);    // 256 MB
  const size_t ZH_BYTES = (size_t)N * D * sizeof(_Float16); // 2 MB
  float*     Kmat = (float*)ws;
  _Float16*  zh   = (_Float16*)(ws + K_BYTES);
  float*     u    = (float*)(ws + K_BYTES + ZH_BYTES);
  float*     v    = u + N;
  float*     nll  = v + N;
  unsigned*  minb = (unsigned*)(nll + N);

  const float invN = 1.0f / (float)N;

  init_k<<<N / 256, 256, 0, stream>>>(u, v, minb, invN);
  normalize_k<<<N, 128, 0, stream>>>(z, zh);
  gemm_scores<<<(256 * 256) / 8, 256, 0, stream>>>(zh, Kmat, minb);
  gibbs_k<<<(int)(((size_t)N * N) / 1024), 256, 0, stream>>>(Kmat, minb);
  for (int it = 0; it < 10; ++it) {      // K is symmetric: K^T u == K u
    matvec_div<<<N, 256, 0, stream>>>(Kmat, v, u, invN);
    matvec_div<<<N, 256, 0, stream>>>(Kmat, u, v, invN);
  }
  loss_rows<<<N, 256, 0, stream>>>(Kmat, v, nll);
  finalize_k<<<1, 256, 0, stream>>>(nll, out);
}
